// KernelNN_52896817218079
// MI455X (gfx1250) — matrix operand order repaired and checked by reading, not compiled, not run on hardware
//
#include <hip/hip_runtime.h>
#include <stdint.h>

#define WIDTH  64
#define KER_W  1024
#define KER_IN 6
#define DEPTH  6

typedef __attribute__((ext_vector_type(16))) __bf16       bf16x16;
typedef __attribute__((ext_vector_type(8)))  float        f32x8;
typedef __attribute__((ext_vector_type(4)))  unsigned int u32x4;
typedef __attribute__((ext_vector_type(4)))  int          v4i;

union Frag {
    bf16x16 v;
    u32x4   q[2];
};

#if defined(__gfx1250__) && __has_builtin(__builtin_amdgcn_global_load_async_to_lds_b128)
#define USE_ASYNC_LDS 1
#else
#define USE_ASYNC_LDS 0
#endif

__device__ __forceinline__ void async_cp16(const unsigned short* g, unsigned short* l) {
#if USE_ASYNC_LDS
    // ROCm 7.2 clang-22 declares this builtin with generic `v4i*` params
    __builtin_amdgcn_global_load_async_to_lds_b128((v4i*)g, (v4i*)l, 0, 0);
#else
    *(u32x4*)l = *(const u32x4*)g;
#endif
}

__device__ __forceinline__ void wait_async_all() {
#if defined(__gfx1250__)
#if __has_builtin(__builtin_amdgcn_s_wait_asynccnt)
    __builtin_amdgcn_s_wait_asynccnt(0);
#else
    asm volatile("s_wait_asynccnt 0x0" ::: "memory");
#endif
#endif
}

__device__ __forceinline__ unsigned short f2bf(float f) {
    unsigned int u = __float_as_uint(f);
    unsigned int r = u + 0x7FFFu + ((u >> 16) & 1u);   // round-to-nearest-even
    return (unsigned short)(r >> 16);
}
__device__ __forceinline__ float bf2f(unsigned short h) {
    return __uint_as_float(((unsigned int)h) << 16);
}

// ---------------------------------------------------------------- utilities
__global__ void zero_f32_kernel(float* __restrict__ p, int n) {
    int i = blockIdx.x * blockDim.x + threadIdx.x;
    if (i < n) p[i] = 0.0f;
}

// w [K x Nn] row-major f32  ->  wt [Nn x K] bf16 (i.e. B^T, K-contiguous)
__global__ void convert_transpose_kernel(const float* __restrict__ w,
                                         unsigned short* __restrict__ wt,
                                         int K, int Nn) {
    int i = blockIdx.x * blockDim.x + threadIdx.x;
    if (i >= K * Nn) return;
    int n = i / K;
    int k = i - n * K;
    wt[i] = f2bf(w[(size_t)k * Nn + n]);
}

// z0[n][o] = x[n]*fc1_w[o] + fc1_b[o]
__global__ void fc1_kernel(const float* __restrict__ x, const float* __restrict__ w,
                           const float* __restrict__ b, float* __restrict__ z, int N) {
    int i = blockIdx.x * blockDim.x + threadIdx.x;
    if (i >= N * WIDTH) return;
    int n = i >> 6, o = i & 63;
    z[i] = x[n] * w[o] + b[o];
}

__global__ void deg_kernel(const int* __restrict__ dst, float* __restrict__ deg, int E) {
    int e = blockIdx.x * blockDim.x + threadIdx.x;
    if (e < E) atomicAdd(&deg[dst[e]], 1.0f);
}

// h1 = relu(edge_attr @ k1_w + k1_b) -> bf16   (K=6, VALU is fine)
__global__ __launch_bounds__(256)
void h1_kernel(const float* __restrict__ ea, const float* __restrict__ k1w,
               const float* __restrict__ k1b, unsigned short* __restrict__ h1, int E) {
    int e = blockIdx.x;
    int c = blockIdx.y * 256 + threadIdx.x;
    if (e >= E) return;
    float acc = k1b[c];
#pragma unroll
    for (int i = 0; i < KER_IN; ++i)
        acc += ea[(size_t)e * KER_IN + i] * k1w[i * KER_W + c];
    h1[(size_t)e * KER_W + c] = f2bf(fmaxf(acc, 0.0f));
}

// --------------------------------------------------- bf16 WMMA GEMM (CDNA5)
// C[M x Nn] = act(A[M x K] @ B + bias),  B given as Bt[Nn x K] (K-contiguous).
// Block = 256 threads = 8 waves; wave tile 32x32 (2x2 v_wmma_f32_16x16x32_bf16);
// block tile 128x64.  K-slices (32 wide) double-buffered in LDS via
// GLOBAL_LOAD_ASYNC_TO_LDS_B128 (ASYNCcnt).  Nn % 64 == 0, K % 32 == 0.
__global__ __launch_bounds__(256)
void wmma_gemm_bf16(const unsigned short* __restrict__ A,
                    const unsigned short* __restrict__ Bt,
                    const float* __restrict__ bias,
                    unsigned short* __restrict__ C,
                    int M, int Nn, int K, int relu) {
    __shared__ unsigned short lsA[2][128 * 32];   // 8 KB per buffer
    __shared__ unsigned short lsB[2][64 * 32];    // 4 KB per buffer

    const int tid  = threadIdx.x;
    const int lane = tid & 31;
    const int wave = tid >> 5;
    const int mBlk = (int)blockIdx.x * 128;
    const int nBlk = (int)blockIdx.y * 64;
    const int m0w  = (wave >> 1) * 32;   // wave tile origin (local)
    const int n0w  = (wave & 1) * 32;

    // --- staging map: A tile = 512 x 16B chunks (2/thread), B tile = 256 (1/thread)
    const int aRow0 = tid >> 2;            // 0..63
    const int aRow1 = aRow0 + 64;          // 64..127
    const int part  = (tid & 3) * 8;       // element offset inside 32-wide K slice
    int gRow0 = mBlk + aRow0; if (gRow0 >= M) gRow0 = M - 1;   // clamp OOB reads
    int gRow1 = mBlk + aRow1; if (gRow1 >= M) gRow1 = M - 1;
    const int bCol = nBlk + (tid >> 2);

    const size_t aOff0 = (size_t)gRow0 * K + part;
    const size_t aOff1 = (size_t)gRow1 * K + part;
    const size_t bOff  = (size_t)bCol * K + part;
    const int ldsA0 = aRow0 * 32 + part;
    const int ldsA1 = aRow1 * 32 + part;
    const int ldsB0 = (tid >> 2) * 32 + part;

    const f32x8 zero = {0.f, 0.f, 0.f, 0.f, 0.f, 0.f, 0.f, 0.f};
    f32x8 acc[2][2];
    acc[0][0] = zero; acc[0][1] = zero; acc[1][0] = zero; acc[1][1] = zero;

    const int nsteps = K >> 5;

    // preload K-slice 0 into buffer 0
    async_cp16(A + aOff0,  &lsA[0][ldsA0]);
    async_cp16(A + aOff1,  &lsA[0][ldsA1]);
    async_cp16(Bt + bOff,  &lsB[0][ldsB0]);

    // WMMA fragment lane map (ISA 16-bit A 16x32 / B 32x16 layouts)
    const int frow = lane & 15;
    const int asub = (lane >> 4) * 8;    // A: K sub-offset {0,8}; 2nd chunk at +16
    const int bsub = (lane >> 4) * 8;    // B: K sub-offset {0,8}; 2nd chunk at +16

    for (int s = 0; s < nsteps; ++s) {
        const int buf = s & 1;
#if USE_ASYNC_LDS
        wait_async_all();                 // buffer `buf` resident in LDS
#endif
        __syncthreads();                  // ...for every wave; buf^1 free to refill
        if (s + 1 < nsteps) {
            const int kk = (s + 1) << 5;
            async_cp16(A + aOff0 + kk, &lsA[buf ^ 1][ldsA0]);
            async_cp16(A + aOff1 + kk, &lsA[buf ^ 1][ldsA1]);
            async_cp16(Bt + bOff + kk, &lsB[buf ^ 1][ldsB0]);
        }
        if (s + 2 < nsteps)               // keep a far prefetch in flight (L2)
            __builtin_prefetch((const void*)(A + aOff0 + ((s + 2) << 5)), 0, 1);

        Frag a[2], b[2];
#pragma unroll
        for (int mi = 0; mi < 2; ++mi) {
            const unsigned short* p = &lsA[buf][(m0w + mi * 16 + frow) * 32 + asub];
            a[mi].q[0] = *(const u32x4*)p;          // K = asub .. asub+7
            a[mi].q[1] = *(const u32x4*)(p + 16);   // K = asub+16 .. asub+23
        }
#pragma unroll
        for (int ni = 0; ni < 2; ++ni) {
            const unsigned short* p = &lsB[buf][(n0w + ni * 16 + frow) * 32 + bsub];
            b[ni].q[0] = *(const u32x4*)p;          // K = bsub .. bsub+7
            b[ni].q[1] = *(const u32x4*)(p + 16);   // K = bsub+16 .. bsub+23
        }
#pragma unroll
        for (int mi = 0; mi < 2; ++mi)
#pragma unroll
            for (int ni = 0; ni < 2; ++ni)
                acc[mi][ni] = __builtin_amdgcn_wmma_f32_16x16x32_bf16(
                    false, a[mi].v, false, b[ni].v, (short)0, acc[mi][ni], false, false);
    }

    // C/D layout: VGPR i -> row i (lanes 0-15) / row 8+i (lanes 16-31), col = lane&15
    const int scol = lane & 15;
    const int srow = (lane >> 4) * 8;
#pragma unroll
    for (int mi = 0; mi < 2; ++mi) {
#pragma unroll
        for (int ni = 0; ni < 2; ++ni) {
            int col = nBlk + n0w + ni * 16 + scol;
            float bv = bias ? bias[col] : 0.0f;
#pragma unroll
            for (int i = 0; i < 8; ++i) {
                int row = mBlk + m0w + mi * 16 + srow + i;
                if (row < M) {
                    float v = acc[mi][ni][i] + bv;
                    if (relu) v = fmaxf(v, 0.0f);
                    C[(size_t)row * Nn + col] = f2bf(v);
                }
            }
        }
    }
}

// ------------------------------------------------- message passing (x DEPTH)
// msg[e][o] = sum_i z[src[e]][i] * ew[e][i][o];  agg[dst[e]][o] += msg
__global__ __launch_bounds__(256)
void edge_msg_kernel(const int* __restrict__ src, const int* __restrict__ dst,
                     const float* __restrict__ z, const unsigned short* __restrict__ ew,
                     float* __restrict__ agg, int E) {
    __shared__ float zs[4][WIDTH];
    const int q = threadIdx.x >> 6;
    const int o = threadIdx.x & 63;
    const int e = blockIdx.x * 4 + q;
    const bool valid = e < E;
    int s = 0, d = 0;
    if (valid) {
        s = src[e];
        d = dst[e];
        zs[q][o] = z[(size_t)s * WIDTH + o];
    }
    __syncthreads();
    if (!valid) return;
    const unsigned short* W = ew + (size_t)e * WIDTH * WIDTH + o;
    float acc = 0.0f;
#pragma unroll 8
    for (int i = 0; i < WIDTH; ++i)
        acc += zs[q][i] * bf2f(W[(size_t)i * WIDTH]);
    atomicAdd(&agg[(size_t)d * WIDTH + o], acc);
}

// z_out = relu(agg/deg + z_in @ root_w + conv_b)
__global__ __launch_bounds__(256)
void node_update_kernel(const float* __restrict__ zin, const float* __restrict__ agg,
                        const float* __restrict__ deg, const float* __restrict__ rootw,
                        const float* __restrict__ convb, float* __restrict__ zout, int N) {
    __shared__ float zs[4][WIDTH];
    const int q = threadIdx.x >> 6;
    const int o = threadIdx.x & 63;
    const int n = blockIdx.x * 4 + q;
    const bool valid = n < N;
    if (valid) zs[q][o] = zin[(size_t)n * WIDTH + o];
    __syncthreads();
    if (!valid) return;
    float dg  = fmaxf(deg[n], 1.0f);
    float acc = agg[(size_t)n * WIDTH + o] / dg + convb[o];
#pragma unroll 8
    for (int i = 0; i < WIDTH; ++i)
        acc += zs[q][i] * rootw[i * WIDTH + o];
    zout[(size_t)n * WIDTH + o] = fmaxf(acc, 0.0f);
}

__global__ void fc2_kernel(const float* __restrict__ z, const float* __restrict__ w,
                           const float* __restrict__ b, float* __restrict__ out, int N) {
    int n = blockIdx.x * blockDim.x + threadIdx.x;
    if (n >= N) return;
    float acc = b[0];
#pragma unroll 8
    for (int o = 0; o < WIDTH; ++o)
        acc += z[(size_t)n * WIDTH + o] * w[o];
    out[n] = acc;
}

// ----------------------------------------------------------------- launcher
extern "C" void kernel_launch(void* const* d_in, const int* in_sizes, int n_in,
                              void* d_out, int out_size, void* d_ws, size_t ws_size,
                              hipStream_t stream) {
    const float* x      = (const float*)d_in[0];
    const int*   ei     = (const int*)  d_in[1];
    const float* ea     = (const float*)d_in[2];
    const float* fc1_w  = (const float*)d_in[3];
    const float* fc1_b  = (const float*)d_in[4];
    const float* k1_w   = (const float*)d_in[5];
    const float* k1_b   = (const float*)d_in[6];
    const float* k2_w   = (const float*)d_in[7];
    const float* k2_b   = (const float*)d_in[8];
    const float* k3_w   = (const float*)d_in[9];
    const float* k3_b   = (const float*)d_in[10];
    const float* root_w = (const float*)d_in[11];
    const float* conv_b = (const float*)d_in[12];
    const float* fc2_w  = (const float*)d_in[13];
    const float* fc2_b  = (const float*)d_in[14];
    float* out = (float*)d_out;

    const int N = in_sizes[0];        // nodes
    const int E = in_sizes[1] / 2;    // edges
    const int* src = ei;
    const int* dst = ei + E;

    // ---- workspace carve-up (256B aligned)
    char*  ws  = (char*)d_ws;
    size_t off = 0;
    auto alloc = [&](size_t bytes) -> void* {
        off = (off + 255) & ~(size_t)255;
        void* p = ws + off;
        off += bytes;
        return p;
    };
    unsigned short* k2t = (unsigned short*)alloc((size_t)KER_W * KER_W * 2);          // [1024][1024] bf16 (B^T)
    unsigned short* k3t = (unsigned short*)alloc((size_t)WIDTH * WIDTH * KER_W * 2);  // [4096][1024] bf16 (B^T)
    unsigned short* h1  = (unsigned short*)alloc((size_t)E * KER_W * 2);              // [E][1024] bf16
    unsigned short* h2  = (unsigned short*)alloc((size_t)E * KER_W * 2);              // [E][1024] bf16
    unsigned short* ew  = (unsigned short*)alloc((size_t)E * WIDTH * WIDTH * 2);      // [E][64][64] bf16
    float* zA  = (float*)alloc((size_t)N * WIDTH * 4);
    float* zB  = (float*)alloc((size_t)N * WIDTH * 4);
    float* agg = (float*)alloc((size_t)N * WIDTH * 4);
    float* deg = (float*)alloc((size_t)N * 4);

    // ---- one-time prep
    {
        int tot = KER_W * KER_W;
        convert_transpose_kernel<<<(tot + 255) / 256, 256, 0, stream>>>(k2_w, k2t, KER_W, KER_W);
        tot = KER_W * WIDTH * WIDTH;
        convert_transpose_kernel<<<(tot + 255) / 256, 256, 0, stream>>>(k3_w, k3t, KER_W, WIDTH * WIDTH);
    }
    fc1_kernel<<<(N * WIDTH + 255) / 256, 256, 0, stream>>>(x, fc1_w, fc1_b, zA, N);
    zero_f32_kernel<<<(N + 255) / 256, 256, 0, stream>>>(deg, N);
    deg_kernel<<<(E + 255) / 256, 256, 0, stream>>>(dst, deg, E);

    // ---- kernel MLP: h1 -> h2 -> ew  (WMMA GEMMs, LDS async double-buffered)
    {
        dim3 g1(E, KER_W / 256);
        h1_kernel<<<g1, 256, 0, stream>>>(ea, k1_w, k1_b, h1, E);

        dim3 g2((E + 127) / 128, KER_W / 64);
        wmma_gemm_bf16<<<g2, 256, 0, stream>>>(h1, k2t, k2_b, h2, E, KER_W, KER_W, 1);

        dim3 g3((E + 127) / 128, (WIDTH * WIDTH) / 64);
        wmma_gemm_bf16<<<g3, 256, 0, stream>>>(h2, k3t, k3_b, ew, E, WIDTH * WIDTH, KER_W, 0);
    }

    // ---- DEPTH message-passing iterations (ping-pong z)
    float* zin  = zA;
    float* zout = zB;
    for (int it = 0; it < DEPTH; ++it) {
        zero_f32_kernel<<<(N * WIDTH + 255) / 256, 256, 0, stream>>>(agg, N * WIDTH);
        edge_msg_kernel<<<(E + 3) / 4, 256, 0, stream>>>(src, dst, zin, ew, agg, E);
        node_update_kernel<<<(N + 3) / 4, 256, 0, stream>>>(zin, agg, deg, root_w, conv_b, zout, N);
        float* t = zin; zin = zout; zout = t;
    }

    // ---- readout (final z is in `zin` after the swap)
    fc2_kernel<<<(N + 255) / 256, 256, 0, stream>>>(zin, fc2_w, fc2_b, out, N);
}
